// AttSeq_26620207300628
// MI455X (gfx1250) — compile-verified, matrix operand order checked by reading, not run on hardware
//
#include <hip/hip_runtime.h>
#include <hip/hip_bf16.h>
#include <math.h>

// ---------------------------------------------------------------------------
// Types for CDNA5 WMMA (wave32): A/B = 16 bf16 per lane, C/D = 8 f32 per lane
// ---------------------------------------------------------------------------
typedef __attribute__((ext_vector_type(16))) __bf16 v16bf;
typedef __attribute__((ext_vector_type(8)))  float  v8f;

union FragAB { v16bf v; unsigned int w[8]; };

#define NDIM 256   // all GEMMs here have N = 256
#define TM   64    // block M tile (2 m-tiles per wave, 2 wave M-groups)
#define KT   32    // K step per WMMA (16x16x32 bf16)
#define LDSK 34    // padded K stride (bf16 elems) -> 17 dwords, conflict-free

// Fragment load (ISA 16-bit layout: k = (r<4?0:16) + 8*half + 2*(r&3)) + 8 WMMAs
__device__ __forceinline__ void mma_step(const __bf16* __restrict__ Als,
                                         const __bf16* __restrict__ Bls,
                                         int wm, int wn, int half, int l16,
                                         v8f (&acc)[2][4])
{
    const unsigned int* Aw = (const unsigned int*)Als;
    const unsigned int* Bw = (const unsigned int*)Bls;

    FragAB afr[2];
    #pragma unroll
    for (int mt = 0; mt < 2; ++mt) {
        int m = wm * 32 + mt * 16 + l16;
        #pragma unroll
        for (int r = 0; r < 8; ++r) {
            int k = ((r < 4) ? 0 : 16) + 8 * half + 2 * (r & 3);
            afr[mt].w[r] = Aw[m * (LDSK / 2) + (k >> 1)];
        }
    }
    #pragma unroll
    for (int nt = 0; nt < 4; ++nt) {
        FragAB bfr;
        int n = wn * 64 + nt * 16 + l16;
        #pragma unroll
        for (int r = 0; r < 8; ++r) {
            int k = ((r < 4) ? 0 : 16) + 8 * half + 2 * (r & 3);
            bfr.w[r] = Bw[n * (LDSK / 2) + (k >> 1)];
        }
        #pragma unroll
        for (int mt = 0; mt < 2; ++mt) {
            acc[mt][nt] = __builtin_amdgcn_wmma_f32_16x16x32_bf16(
                false, afr[mt].v, false, bfr.v,
                (short)0, acc[mt][nt], false, false);
        }
    }
}

// ---------------------------------------------------------------------------
// GEMM: C[M x 256] = act( rowscale(A) [M x K] @ B [K x 256] + bias ) + resid
//   rowscale/bias/resid nullable; act: 0 = identity, 1 = relu
//   A row-major (lda), B row-major (ldb = 256). bf16 WMMA, f32 accumulate.
//   8 waves arranged 2(M) x 4(N); each wave: 2 m-tiles x 4 n-tiles = 8 accs.
//   Staging is fully branch-free: OOB rows clamp to row 0 with a zero row
//   scale; K main loop is whole steps only, one masked tail step if needed.
// ---------------------------------------------------------------------------
__global__ void __launch_bounds__(256)
gemm_wmma_n256(const float* __restrict__ A, int lda,
               const float* __restrict__ B,
               const float* __restrict__ rowscale,
               const float* __restrict__ bias,
               const float* __restrict__ resid,
               float* __restrict__ C,
               int M, int K, int act)
{
    __shared__ __bf16 Als[TM  * LDSK];   // [m][k]
    __shared__ __bf16 Bls[NDIM * LDSK];  // [n][k] (B transposed at staging)

    const int tid  = threadIdx.x;
    const int lane = tid & 31;
    const int wv   = tid >> 5;     // 0..7
    const int wm   = wv & 1;       // wave M group: rows wm*32 .. +31
    const int wn   = wv >> 1;      // wave N group: cols wn*64 .. +63
    const int m0   = blockIdx.x * TM;
    const int half = lane >> 4;
    const int l16  = lane & 15;

    // staging thread assignments (contiguous-memory ownership)
    const int ar  = tid >> 2;          // A: row 0..63
    const int akq = (tid & 3) * 8;     // A: k offset 0,8,16,24
    const int bk  = tid >> 3;          // B: k row 0..31
    const int bn0 = (tid & 7) * 32;    // B: n offset 0..224

    // loop-invariant A-row state: clamp OOB rows to row 0, fold mask into scale
    const int  gr_a   = m0 + ar;
    const bool arowok = gr_a < M;
    const float rsA   = arowok ? (rowscale ? rowscale[gr_a] : 1.0f) : 0.0f;
    const float* __restrict__ arow = A + (size_t)(arowok ? gr_a : 0) * lda;

    v8f acc[2][4];
    for (int i = 0; i < 2; ++i)
        for (int j = 0; j < 4; ++j)
            for (int q = 0; q < 8; ++q) acc[i][j][q] = 0.0f;

    const int Kmain = K & ~(KT - 1);

    for (int k0 = 0; k0 < Kmain; k0 += KT) {
        // ---- stage A tile (64 x 32): two aligned float4 loads, no checks ----
        {
            __bf16* dst = &Als[ar * LDSK + akq];
            const float4* src = (const float4*)(arow + k0 + akq);
            float4 p0 = src[0];
            float4 p1 = src[1];
            dst[0] = (__bf16)(p0.x * rsA); dst[1] = (__bf16)(p0.y * rsA);
            dst[2] = (__bf16)(p0.z * rsA); dst[3] = (__bf16)(p0.w * rsA);
            dst[4] = (__bf16)(p1.x * rsA); dst[5] = (__bf16)(p1.y * rsA);
            dst[6] = (__bf16)(p1.z * rsA); dst[7] = (__bf16)(p1.w * rsA);
        }
        // ---- stage B tile (32 x 256) transposed to [n][k], no checks ----
        {
            const float4* src = (const float4*)(B + (size_t)(k0 + bk) * NDIM + bn0);
            #pragma unroll
            for (int q = 0; q < 8; ++q) {
                float4 p = src[q];
                int n = bn0 + q * 4;
                Bls[(n + 0) * LDSK + bk] = (__bf16)p.x;
                Bls[(n + 1) * LDSK + bk] = (__bf16)p.y;
                Bls[(n + 2) * LDSK + bk] = (__bf16)p.z;
                Bls[(n + 3) * LDSK + bk] = (__bf16)p.w;
            }
        }
        __syncthreads();
        mma_step(Als, Bls, wm, wn, half, l16, acc);
        __syncthreads();
    }

    // ---- masked K tail (only when K % 32 != 0, e.g. K = 300) ----
    if (Kmain < K) {
        const int k0 = Kmain;
        {
            __bf16* dst = &Als[ar * LDSK + akq];
            #pragma unroll
            for (int j = 0; j < 8; ++j) {
                int gk = k0 + akq + j;
                int ck = (gk < K) ? gk : (K - 1);            // clamped valid load
                float v = arow[ck] * ((gk < K) ? rsA : 0.0f);
                dst[j] = (__bf16)v;
            }
        }
        {
            const int gk = k0 + bk;
            const bool kok = gk < K;
            const float* __restrict__ brow = B + (size_t)(kok ? gk : (K - 1)) * NDIM;
            const float km = kok ? 1.0f : 0.0f;
            #pragma unroll
            for (int j = 0; j < 32; ++j)
                Bls[(bn0 + j) * LDSK + bk] = (__bf16)(brow[bn0 + j] * km);
        }
        __syncthreads();
        mma_step(Als, Bls, wm, wn, half, l16, acc);
        __syncthreads();
    }

    // ---- fused epilogue: bias -> act -> +resid -> store ----
    for (int mt = 0; mt < 2; ++mt) {
        for (int nt = 0; nt < 4; ++nt) {
            int n = wn * 64 + nt * 16 + l16;
            #pragma unroll
            for (int r = 0; r < 8; ++r) {
                int m  = wm * 32 + mt * 16 + r + 8 * half;  // C layout: VGPR r -> M
                int gr = m0 + m;
                if (gr < M) {
                    float t = acc[mt][nt][r];
                    if (bias) t += bias[n];
                    if (act == 1) t = fmaxf(t, 0.0f);
                    if (resid) t += resid[(size_t)gr * NDIM + n];
                    C[(size_t)gr * NDIM + n] = t;
                }
            }
        }
    }
}

// ---------------------------------------------------------------------------
// Small helper kernels
// ---------------------------------------------------------------------------
__global__ void zero_k(float* __restrict__ p, size_t n) {
    size_t i = blockIdx.x * (size_t)blockDim.x + threadIdx.x;
    if (i < n) p[i] = 0.0f;
}

// out[row] = dot(X[row, 0:K], w[0:K]); one wave per row
__global__ void __launch_bounds__(256)
rowdot_k(const float* __restrict__ X, const float* __restrict__ w,
         float* __restrict__ out, int M, int K)
{
    int wid  = blockIdx.x * 8 + (threadIdx.x >> 5);
    int lane = threadIdx.x & 31;
    if (wid >= M) return;
    float s = 0.0f;
    for (int k = lane; k < K; k += 32) s += X[(size_t)wid * K + k] * w[k];
    for (int off = 16; off > 0; off >>= 1) s += __shfl_xor(s, off, 32);
    if (lane == 0) out[wid] = s;
}

// per-edge attention score: exp(leaky_relu(u[h] + t[att] + b)); rowsum via atomics
__global__ void edge_score_k(const int* __restrict__ tr,
                             const float* __restrict__ u,
                             const float* __restrict__ tatt,
                             const float* __restrict__ ab,
                             float* __restrict__ esc,
                             float* __restrict__ rowsum, int E)
{
    int e = blockIdx.x * blockDim.x + threadIdx.x;
    if (e >= E) return;
    int h = tr[3 * e], att = tr[3 * e + 2];
    float s = u[h] + tatt[att] + ab[0];
    s = (s > 0.0f) ? s : 0.2f * s;
    float sc = expf(s);
    esc[e] = sc;
    atomicAdd(&rowsum[h], sc);
}

// agg[h] += p_e * (att_proj[att] + val_proj[val]); one wave per edge
__global__ void __launch_bounds__(256)
edge_agg_k(const int* __restrict__ tr,
           const float* __restrict__ esc,
           const float* __restrict__ rowsum,
           const float* __restrict__ attp,
           const float* __restrict__ valp,
           float* __restrict__ agg, int E)
{
    int wid  = blockIdx.x * 8 + (threadIdx.x >> 5);
    int lane = threadIdx.x & 31;
    if (wid >= E) return;
    int h = tr[3 * wid], val = tr[3 * wid + 1], att = tr[3 * wid + 2];
    float p = esc[wid] / rowsum[h];
    #pragma unroll
    for (int i = 0; i < 8; ++i) {
        int d = lane + i * 32;
        float av = attp[(size_t)att * NDIM + d] + valp[(size_t)val * NDIM + d];
        atomicAdd(&agg[(size_t)h * NDIM + d], p * av);
    }
}

// out = elu(agg + ent)
__global__ void enc_finish_k(const float* __restrict__ agg,
                             const float* __restrict__ ent,
                             float* __restrict__ out, size_t n)
{
    size_t i = blockIdx.x * (size_t)blockDim.x + threadIdx.x;
    if (i >= n) return;
    float v = agg[i] + ent[i];
    out[i] = (v > 0.0f) ? v : (expf(v) - 1.0f);
}

// GCN scatter: agg[rows[e]] += feats[cols[e]]; deg[rows[e]] += 1; wave per edge
__global__ void __launch_bounds__(256)
gcn_scatter_k(const int* __restrict__ rows, const int* __restrict__ cols,
              const float* __restrict__ feats,
              float* __restrict__ agg, float* __restrict__ deg, int E)
{
    int wid  = blockIdx.x * 8 + (threadIdx.x >> 5);
    int lane = threadIdx.x & 31;
    if (wid >= E) return;
    int r = rows[wid], c = cols[wid];
    if (lane == 0) atomicAdd(&deg[r], 1.0f);
    #pragma unroll
    for (int i = 0; i < 8; ++i) {
        int d = lane + i * 32;
        atomicAdd(&agg[(size_t)r * NDIM + d], feats[(size_t)c * NDIM + d]);
    }
}

__global__ void recip_k(float* __restrict__ p, int n) {
    int i = blockIdx.x * blockDim.x + threadIdx.x;
    if (i >= n) return;
    float d = p[i];
    p[i] = (d > 0.0f) ? (1.0f / d) : 0.0f;
}

// dst[row] = src[row] / max(||src[row]||, 1e-12); one wave per row
__global__ void __launch_bounds__(256)
l2norm_k(const float* __restrict__ src, float* __restrict__ dst, int M)
{
    int wid  = blockIdx.x * 8 + (threadIdx.x >> 5);
    int lane = threadIdx.x & 31;
    if (wid >= M) return;
    float ss = 0.0f;
    #pragma unroll
    for (int i = 0; i < 8; ++i) {
        float v = src[(size_t)wid * NDIM + lane + i * 32];
        ss += v * v;
    }
    for (int off = 16; off > 0; off >>= 1) ss += __shfl_xor(ss, off, 32);
    float sc = 1.0f / fmaxf(sqrtf(ss), 1e-12f);
    #pragma unroll
    for (int i = 0; i < 8; ++i) {
        int d = lane + i * 32;
        dst[(size_t)wid * NDIM + d] = src[(size_t)wid * NDIM + d] * sc;
    }
}

__global__ void gather_k(const float* __restrict__ src, const int* __restrict__ idx,
                         float* __restrict__ dst, size_t n)
{
    size_t i = blockIdx.x * (size_t)blockDim.x + threadIdx.x;
    if (i >= n) return;
    size_t row = i >> 8;
    int d = (int)(i & 255);
    dst[i] = src[(size_t)idx[row] * NDIM + d];
}

// ---------------------------------------------------------------------------
// Host orchestration
// ---------------------------------------------------------------------------
static inline int cdiv_i(long long a, long long b) { return (int)((a + b - 1) / b); }

extern "C" void kernel_launch(void* const* d_in, const int* in_sizes, int n_in,
                              void* d_out, int out_size, void* d_ws, size_t ws_size,
                              hipStream_t stream)
{
    const int*   seed_sr  = (const int*)d_in[0];
    const int*   seed_tg  = (const int*)d_in[1];
    const int*   tri_sr   = (const int*)d_in[2];
    const int*   tri_tg   = (const int*)d_in[3];
    const int*   rows_sr  = (const int*)d_in[4];
    const int*   cols_sr  = (const int*)d_in[5];
    const int*   rows_tg  = (const int*)d_in[6];
    const int*   cols_tg  = (const int*)d_in[7];
    const float* attf     = (const float*)d_in[8];
    const float* valf     = (const float*)d_in[9];
    const float* ent_sr   = (const float*)d_in[10];
    const float* ent_tg   = (const float*)d_in[11];
    const float* a_w      = (const float*)d_in[12];
    const float* a_b      = (const float*)d_in[13];
    const float* W_enc    = (const float*)d_in[14];
    const float* w1       = (const float*)d_in[15];
    const float* b1       = (const float*)d_in[16];
    const float* w2       = (const float*)d_in[17];
    const float* b2       = (const float*)d_in[18];

    const int NSEED_SR = in_sizes[0];
    const int NSEED_TG = in_sizes[1];
    const int EA_SR    = in_sizes[2] / 3;
    const int EA_TG    = in_sizes[3] / 3;
    const int EG_SR    = in_sizes[4];
    const int EG_TG    = in_sizes[6];
    const int ATT      = in_sizes[8] / NDIM;
    const int VAL_DIM  = 300;
    const int NVAL     = in_sizes[9] / VAL_DIM;
    const int NSR      = in_sizes[10] / NDIM;
    const int NTG      = in_sizes[11] / NDIM;
    const int NMAX     = (NSR > NTG) ? NSR : NTG;
    const int EAMAX    = (EA_SR > EA_TG) ? EA_SR : EA_TG;

    // ---- carve workspace ----
    size_t off = 0;
    auto wsAlloc = [&](size_t elems) -> float* {
        float* p = (float*)((char*)d_ws + off);
        off += ((elems * sizeof(float)) + 255) & ~(size_t)255;
        return p;
    };
    float* val_proj = wsAlloc((size_t)NVAL * NDIM);   // val_feats @ W_enc[256:]
    float* att_proj = wsAlloc((size_t)ATT  * NDIM);   // att_feats @ W_enc[:256]
    float* fs       = wsAlloc((size_t)NSR  * NDIM);
    float* ft       = wsAlloc((size_t)NTG  * NDIM);
    float* tmp      = wsAlloc((size_t)NMAX * NDIM);   // agg scratch
    float* u_sr     = wsAlloc((size_t)NSR);
    float* u_tg     = wsAlloc((size_t)NTG);
    float* t_att    = wsAlloc((size_t)ATT);
    float* rvec     = wsAlloc((size_t)NMAX);          // rowsum / deg
    float* esc      = wsAlloc((size_t)EAMAX);
    (void)ws_size; (void)n_in;

    // ---- pre-projections (the big factorization win) ----
    gemm_wmma_n256<<<cdiv_i(ATT, TM), 256, 0, stream>>>(
        attf, NDIM, W_enc, nullptr, nullptr, nullptr, att_proj, ATT, NDIM, 0);
    gemm_wmma_n256<<<cdiv_i(NVAL, TM), 256, 0, stream>>>(
        valf, VAL_DIM, W_enc + (size_t)NDIM * NDIM, nullptr, nullptr, nullptr,
        val_proj, NVAL, VAL_DIM, 0);

    // ---- attention score pre-dots ----
    rowdot_k<<<cdiv_i(NSR, 8), 256, 0, stream>>>(ent_sr, a_w, u_sr, NSR, NDIM);
    rowdot_k<<<cdiv_i(NTG, 8), 256, 0, stream>>>(ent_tg, a_w, u_tg, NTG, NDIM);
    rowdot_k<<<cdiv_i(ATT, 8), 256, 0, stream>>>(attf, a_w + NDIM, t_att, ATT, NDIM);

    // ---- attributed encoder (per side) ----
    auto encoder = [&](const int* tri, const float* u, const float* ent,
                       float* f, int N, int E) {
        zero_k<<<cdiv_i(N, 256), 256, 0, stream>>>(rvec, (size_t)N);
        edge_score_k<<<cdiv_i(E, 256), 256, 0, stream>>>(tri, u, t_att, a_b, esc, rvec, E);
        zero_k<<<cdiv_i((long long)N * NDIM, 256), 256, 0, stream>>>(tmp, (size_t)N * NDIM);
        edge_agg_k<<<cdiv_i(E, 8), 256, 0, stream>>>(tri, esc, rvec, att_proj, val_proj, tmp, E);
        enc_finish_k<<<cdiv_i((long long)N * NDIM, 256), 256, 0, stream>>>(
            tmp, ent, f, (size_t)N * NDIM);
    };
    encoder(tri_sr, u_sr, ent_sr, fs, NSR, EA_SR);
    encoder(tri_tg, u_tg, ent_tg, ft, NTG, EA_TG);

    // ---- GCN layer: f = f + act((agg/deg) @ w + b), fully fused GEMM epilogue ----
    auto gcn = [&](const int* rows, const int* cols, float* f, int N, int E,
                   const float* w, const float* b, int act) {
        zero_k<<<cdiv_i((long long)N * NDIM, 256), 256, 0, stream>>>(tmp, (size_t)N * NDIM);
        zero_k<<<cdiv_i(N, 256), 256, 0, stream>>>(rvec, (size_t)N);
        gcn_scatter_k<<<cdiv_i(E, 8), 256, 0, stream>>>(rows, cols, f, tmp, rvec, E);
        recip_k<<<cdiv_i(N, 256), 256, 0, stream>>>(rvec, N);
        gemm_wmma_n256<<<cdiv_i(N, TM), 256, 0, stream>>>(
            tmp, NDIM, w, rvec, b, f, f, N, NDIM, act);
    };
    gcn(rows_sr, cols_sr, fs, NSR, EG_SR, w1, b1, 1);
    gcn(rows_tg, cols_tg, ft, NTG, EG_TG, w1, b1, 1);
    gcn(rows_sr, cols_sr, fs, NSR, EG_SR, w2, b2, 0);
    gcn(rows_tg, cols_tg, ft, NTG, EG_TG, w2, b2, 0);

    // ---- l2norm directly into output segments, then seed gathers ----
    float* out    = (float*)d_out;
    float* out_s0 = out;                                        // fs[seed_sr]
    float* out_s1 = out + (size_t)NSEED_SR * NDIM;              // ft[seed_tg]
    float* out_fs = out + (size_t)(NSEED_SR + NSEED_TG) * NDIM; // fs
    float* out_ft = out_fs + (size_t)NSR * NDIM;                // ft
    (void)out_size;

    l2norm_k<<<cdiv_i(NSR, 8), 256, 0, stream>>>(fs, out_fs, NSR);
    l2norm_k<<<cdiv_i(NTG, 8), 256, 0, stream>>>(ft, out_ft, NTG);
    gather_k<<<cdiv_i((long long)NSEED_SR * NDIM, 256), 256, 0, stream>>>(
        out_fs, seed_sr, out_s0, (size_t)NSEED_SR * NDIM);
    gather_k<<<cdiv_i((long long)NSEED_TG * NDIM, 256), 256, 0, stream>>>(
        out_ft, seed_tg, out_s1, (size_t)NSEED_TG * NDIM);
}
